// GPT_80951543595570
// MI455X (gfx1250) — compile-verified, matrix operand order checked by reading, not run on hardware
//
#include <hip/hip_runtime.h>

typedef _Float16 half_t;
typedef __attribute__((ext_vector_type(16))) _Float16 v16h;
typedef __attribute__((ext_vector_type(8)))  _Float16 v8h;
typedef __attribute__((ext_vector_type(8)))  float    v8f;
typedef __attribute__((ext_vector_type(4)))  unsigned int v4u;
typedef __attribute__((ext_vector_type(4)))  int v4i;

#define GEMM_OUT_F32      0
#define GEMM_OUT_F16      1
#define GEMM_OUT_F16_RELU 2
#define GEMM_ACC_F32      3

// LDS row pitches (halfs). 40 halfs = 80 B: keeps 16-byte alignment of every
// 8-half chunk and spreads rows across banks.
#define PA 40
#define PB 40

#if __has_builtin(__builtin_amdgcn_global_load_async_to_lds_b128) && \
    __has_builtin(__builtin_amdgcn_s_wait_asynccnt)
#define USE_ASYNC_LDS 1
#else
#define USE_ASYNC_LDS 0
#endif

#if USE_ASYNC_LDS
typedef __attribute__((address_space(1))) v4i gas_v4i;  // global int4
typedef __attribute__((address_space(3))) v4i las_v4i;  // LDS int4
#endif

// 16-byte global -> LDS copy (async-to-LDS when the toolchain exposes it).
__device__ inline void g2l_b128(const half_t* __restrict__ g, half_t* l)
{
#if USE_ASYNC_LDS
    __builtin_amdgcn_global_load_async_to_lds_b128(
        (gas_v4i*)(void*)g, (las_v4i*)(void*)l, 0, 0);
#else
    *(v4u*)l = *(const v4u*)g;
#endif
}

__device__ inline void stage_join()
{
#if USE_ASYNC_LDS
    __builtin_amdgcn_s_wait_asynccnt(0);
#endif
    __syncthreads();
}

// ---------------------------------------------------------------------------
// Batched WMMA GEMM: D = A(f16) x B(f16) [+bias], f32 accumulate.
// Block = 128 threads (4 waves). Tile BM=128 x BN=64, BK=32.
// Wave w: rows (w>>1)*64 .. +63, cols (w&1)*32 .. +31  ->  4x2 fragments,
// 8 v_wmma_f32_16x16x32_f16 per K-step, A/B tiles staged in LDS.
// Batch z decomposes as (b,h) with independent base strides (per-head GEMMs).
// Fragment layouts per CDNA5 ISA 7.12.2:
//   A 16x32 f16: lane 0-15 -> M=lane, halfs 0..7 -> K=h, 8..15 -> K=h+8;
//                lanes 16-31 add +8 to K.  B mirrors A with lane -> N.
//   C/D 16x16 f32: vgpr r -> row r (lanes 0-15) / r+8 (lanes 16-31), col=lane&15.
// ---------------------------------------------------------------------------
__global__ __launch_bounds__(128) void wmma_gemm_kernel(
    const half_t* __restrict__ A, const half_t* __restrict__ Bm,
    const float* __restrict__ bias,
    float* __restrict__ Cf, half_t* __restrict__ Ch,
    int M, int N, int K, int transB,
    long lda, long ldb, long ldc,
    long aBase, long aStrB, long aStrH,
    long bBase, long bStrB, long bStrH,
    long cBase, long cStrB, long cStrH,
    int batchH, int mode)
{
    __shared__ __align__(16) half_t Atile[128 * PA];
    __shared__ __align__(16) half_t Btile[64 * PB];

    const int tid  = threadIdx.x;
    const int lane = tid & 31;
    const int wave = tid >> 5;
    const int n0   = blockIdx.x * 64;
    const int m0   = blockIdx.y * 128;
    const int bb   = blockIdx.z;
    const int bIdx = bb / batchH;
    const int hIdx = bb % batchH;
    const long aOff = aBase + (long)bIdx * aStrB + (long)hIdx * aStrH;
    const long bOff = bBase + (long)bIdx * bStrB + (long)hIdx * bStrH;
    const long cOff = cBase + (long)bIdx * cStrB + (long)hIdx * cStrH;

    const int laneLo = lane & 15;
    const int kGrp   = (lane >= 16) ? 8 : 0;  // K half-offset for A/B, row offset for C/D
    const int mbase  = (wave >> 1) * 64;
    const int nbase  = (wave & 1) * 32;

    v8f acc[4][2] = {};

    for (int k0 = 0; k0 < K; k0 += 32) {
        // ---- stage A tile: 128 rows x 32 halfs = 512 chunks of 16 B ----
#pragma unroll
        for (int it = 0; it < 4; ++it) {
            int c  = it * 128 + tid;
            int r  = c >> 2;
            int c8 = (c & 3) << 3;
            const half_t* gp = A + aOff + (long)(m0 + r) * lda + (k0 + c8);
            if (k0 + 32 < K) __builtin_prefetch((const void*)(gp + 32), 0, 1);
            g2l_b128(gp, &Atile[r * PA + c8]);
        }
        // ---- stage B tile as [n][k] ----
        if (transB) {
            // B element (k,n) at bOff + n*ldb + k : contiguous in k -> 16 B chunks
#pragma unroll
            for (int it = 0; it < 2; ++it) {
                int c  = it * 128 + tid;
                int r  = c >> 2;
                int c8 = (c & 3) << 3;
                g2l_b128(Bm + bOff + (long)(n0 + r) * ldb + (k0 + c8),
                         &Btile[r * PB + c8]);
            }
        } else {
            // B element (k,n) at bOff + k*ldb + n : coalesced u16 reads over n,
            // transpose into LDS [n][k]
#pragma unroll
            for (int it = 0; it < 16; ++it) {
                int idx = it * 128 + tid;
                int k   = idx >> 6;
                int n   = idx & 63;
                Btile[n * PB + k] = Bm[bOff + (long)(k0 + k) * ldb + (n0 + n)];
            }
        }
        stage_join();

        // ---- fragments from LDS (two aligned b128 reads each) ----
        v16h af[4], bf[2];
#pragma unroll
        for (int mi = 0; mi < 4; ++mi) {
            const half_t* p = &Atile[(mbase + mi * 16 + laneLo) * PA + kGrp];
            v8h lo = *(const v8h*)p;
            v8h hi = *(const v8h*)(p + 16);
            af[mi] = __builtin_shufflevector(lo, hi, 0, 1, 2, 3, 4, 5, 6, 7,
                                             8, 9, 10, 11, 12, 13, 14, 15);
        }
#pragma unroll
        for (int ni = 0; ni < 2; ++ni) {
            const half_t* p = &Btile[(nbase + ni * 16 + laneLo) * PB + kGrp];
            v8h lo = *(const v8h*)p;
            v8h hi = *(const v8h*)(p + 16);
            bf[ni] = __builtin_shufflevector(lo, hi, 0, 1, 2, 3, 4, 5, 6, 7,
                                             8, 9, 10, 11, 12, 13, 14, 15);
        }

#pragma unroll
        for (int mi = 0; mi < 4; ++mi)
#pragma unroll
            for (int ni = 0; ni < 2; ++ni)
                acc[mi][ni] = __builtin_amdgcn_wmma_f32_16x16x32_f16(
                    false, af[mi], false, bf[ni], (short)0, acc[mi][ni], false, false);

        __syncthreads();   // tiles reused next iteration
    }

    // ---- epilogue ----
#pragma unroll
    for (int mi = 0; mi < 4; ++mi) {
#pragma unroll
        for (int ni = 0; ni < 2; ++ni) {
            const int n = n0 + nbase + ni * 16 + laneLo;
            const float bv = bias ? bias[n] : 0.0f;
#pragma unroll
            for (int r = 0; r < 8; ++r) {
                const int m  = m0 + mbase + mi * 16 + r + kGrp;
                const long ci = cOff + (long)m * ldc + n;
                float v = acc[mi][ni][r] + bv;
                if      (mode == GEMM_OUT_F32)      Cf[ci]  = v;
                else if (mode == GEMM_OUT_F16)      Ch[ci]  = (half_t)v;
                else if (mode == GEMM_OUT_F16_RELU) Ch[ci]  = (half_t)(v > 0.0f ? v : 0.0f);
                else                                Cf[ci] += v;   // residual accumulate
            }
        }
    }
}

// ---------------------------------------------------------------------------
// x[b,t,:] = tok_emb[idx[b,t],:] + pos_emb[t,:]
// ---------------------------------------------------------------------------
__global__ __launch_bounds__(256) void embed_kernel(
    const int* __restrict__ idx, const float* __restrict__ tok,
    const float* __restrict__ pos, float* __restrict__ X, int T, int E)
{
    const long row = blockIdx.x;
    const int  t   = (int)(row % T);
    const int  tk  = idx[row];
    const float* te = tok + (long)tk * E;
    const float* pe = pos + (long)t  * E;
    float* x = X + row * E;
    for (int i = threadIdx.x; i < E; i += blockDim.x) x[i] = te[i] + pe[i];
}

// ---------------------------------------------------------------------------
// LayerNorm over E, f32 in -> f16 out (feeds WMMA GEMMs).
// ---------------------------------------------------------------------------
__global__ __launch_bounds__(256) void layernorm_kernel(
    const float* __restrict__ X, const float* __restrict__ g,
    const float* __restrict__ b, half_t* __restrict__ Y, int E)
{
    __shared__ float red[256];
    const long row = blockIdx.x;
    const float* x = X + row * E;
    const int t = threadIdx.x;

    float s = 0.0f;
    for (int i = t; i < E; i += 256) s += x[i];
    red[t] = s; __syncthreads();
    for (int o = 128; o > 0; o >>= 1) { if (t < o) red[t] += red[t + o]; __syncthreads(); }
    const float mean = red[0] / E;
    __syncthreads();

    float vs = 0.0f;
    for (int i = t; i < E; i += 256) { float d = x[i] - mean; vs += d * d; }
    red[t] = vs; __syncthreads();
    for (int o = 128; o > 0; o >>= 1) { if (t < o) red[t] += red[t + o]; __syncthreads(); }
    const float rstd = rsqrtf(red[0] / E + 1e-5f);

    half_t* y = Y + row * E;
    for (int i = t; i < E; i += 256)
        y[i] = (half_t)((x[i] - mean) * rstd * g[i] + b[i]);
}

// ---------------------------------------------------------------------------
// Causal softmax: one block per (b,h,q) row; 1/sqrt(HS) scale + causal mask,
// writes f16 probabilities for the P x V WMMA GEMM.
// ---------------------------------------------------------------------------
__global__ __launch_bounds__(256) void softmax_kernel(
    const float* __restrict__ S, half_t* __restrict__ P, int T, float scale)
{
    __shared__ float red[256];
    const long row = blockIdx.x;          // row = (b*H + h)*T + q
    const int  q   = (int)(row % T);
    const float* s = S + row * (long)T;
    half_t*      p = P + row * (long)T;
    const int t = threadIdx.x;

    float mx = -1e30f;
    for (int i = t; i < T; i += 256) {
        float v = (i <= q) ? s[i] * scale : -1e9f;
        mx = fmaxf(mx, v);
    }
    red[t] = mx; __syncthreads();
    for (int o = 128; o > 0; o >>= 1) { if (t < o) red[t] = fmaxf(red[t], red[t + o]); __syncthreads(); }
    mx = red[0];
    __syncthreads();

    float sum = 0.0f;
    for (int i = t; i < T; i += 256)
        if (i <= q) sum += expf(s[i] * scale - mx);
    red[t] = sum; __syncthreads();
    for (int o = 128; o > 0; o >>= 1) { if (t < o) red[t] += red[t + o]; __syncthreads(); }
    const float inv = 1.0f / red[0];

    for (int i = t; i < T; i += 256)
        p[i] = (half_t)((i <= q) ? expf(s[i] * scale - mx) * inv : 0.0f);
}

// ---------------------------------------------------------------------------
// f32 -> f16 conversion (weight staging).
// ---------------------------------------------------------------------------
__global__ __launch_bounds__(256) void f32_to_f16_kernel(
    const float* __restrict__ x, half_t* __restrict__ y, long n)
{
    long i = (long)blockIdx.x * 256 + threadIdx.x;
    if (i < n) y[i] = (half_t)x[i];
}

__global__ void zero_kernel(float* p) { if (threadIdx.x == 0 && blockIdx.x == 0) *p = 0.0f; }

// ---------------------------------------------------------------------------
// Cross-entropy: one block per (b,t) row over V logits; mean via atomicAdd.
// ---------------------------------------------------------------------------
__global__ __launch_bounds__(256) void xent_kernel(
    const float* __restrict__ logits, const int* __restrict__ tgt,
    float* __restrict__ loss, int V, float invN)
{
    __shared__ float red[256];
    const long row = blockIdx.x;
    const float* lg = logits + row * (long)V;
    const int t = threadIdx.x;

    float mx = -1e30f;
    for (int i = t; i < V; i += 256) mx = fmaxf(mx, lg[i]);
    red[t] = mx; __syncthreads();
    for (int o = 128; o > 0; o >>= 1) { if (t < o) red[t] = fmaxf(red[t], red[t + o]); __syncthreads(); }
    mx = red[0];
    __syncthreads();

    float sum = 0.0f;
    for (int i = t; i < V; i += 256) sum += expf(lg[i] - mx);
    red[t] = sum; __syncthreads();
    for (int o = 128; o > 0; o >>= 1) { if (t < o) red[t] += red[t + o]; __syncthreads(); }

    if (t == 0) {
        const float lse = mx + logf(red[0]);
        atomicAdd(loss, (lse - lg[tgt[row]]) * invN);
    }
}

// ---------------------------------------------------------------------------
// Host-side orchestration
// ---------------------------------------------------------------------------
static inline void launch_gemm(hipStream_t st,
    const half_t* A, const half_t* B, const float* bias,
    float* Cf, half_t* Ch, int M, int N, int K, int transB,
    long lda, long ldb, long ldc,
    long aB, long aSb, long aSh,
    long bB, long bSb, long bSh,
    long cB, long cSb, long cSh,
    int batB, int batH, int mode)
{
    dim3 grid(N / 64, M / 128, batB * batH);
    wmma_gemm_kernel<<<grid, 128, 0, st>>>(A, B, bias, Cf, Ch, M, N, K, transB,
                                           lda, ldb, ldc, aB, aSb, aSh,
                                           bB, bSb, bSh, cB, cSb, cSh, batH, mode);
}

static inline void launch_cvt(hipStream_t st, const float* x, half_t* y, long n) {
    long blocks = (n + 255) / 256;
    f32_to_f16_kernel<<<(unsigned)blocks, 256, 0, st>>>(x, y, n);
}

extern "C" void kernel_launch(void* const* d_in, const int* in_sizes, int n_in,
                              void* d_out, int out_size, void* d_ws, size_t ws_size,
                              hipStream_t stream)
{
    (void)in_sizes; (void)n_in; (void)out_size;
    const int V = 32000, E = 1024, H = 16, T = 1024, L = 8, Bb = 4;
    const int FF = 4 * E, HS = E / H, BT = Bb * T;

    const int*   idx     = (const int*)  d_in[0];
    const int*   targets = (const int*)  d_in[1];
    const float* tok_emb = (const float*)d_in[2];
    const float* pos_emb = (const float*)d_in[3];
    const float* Wq      = (const float*)d_in[4];
    const float* Wk      = (const float*)d_in[5];
    const float* Wv      = (const float*)d_in[6];
    const float* Wo      = (const float*)d_in[7];
    const float* bo      = (const float*)d_in[8];
    const float* ln1_g   = (const float*)d_in[9];
    const float* ln1_b   = (const float*)d_in[10];
    const float* ln2_g   = (const float*)d_in[11];
    const float* ln2_b   = (const float*)d_in[12];
    const float* W1      = (const float*)d_in[13];
    const float* b1      = (const float*)d_in[14];
    const float* W2      = (const float*)d_in[15];
    const float* b2      = (const float*)d_in[16];
    const float* lnf_g   = (const float*)d_in[17];
    const float* lnf_b   = (const float*)d_in[18];
    const float* Wout    = (const float*)d_in[19];
    const float* bout    = (const float*)d_in[20];

    // ---- workspace layout ----
    char* ws = (char*)d_ws;
    size_t off = 0;
    auto salloc = [&](size_t bytes) -> char* {
        char* p = ws + off;
        off += (bytes + 255) & ~(size_t)255;
        return p;
    };
    float*  X    = (float*) salloc((size_t)BT * E  * sizeof(float));   // residual stream
    half_t* H16  = (half_t*)salloc((size_t)BT * E  * sizeof(half_t));  // LN output
    half_t* Q16  = (half_t*)salloc((size_t)BT * E  * sizeof(half_t));
    half_t* K16  = (half_t*)salloc((size_t)BT * E  * sizeof(half_t));
    half_t* V16  = (half_t*)salloc((size_t)BT * E  * sizeof(half_t));
    float*  S    = (float*) salloc((size_t)Bb * H * T * T * sizeof(float));
    half_t* P16  = (half_t*)salloc((size_t)Bb * H * T * T * sizeof(half_t));
    half_t* O16  = (half_t*)salloc((size_t)BT * E  * sizeof(half_t));
    half_t* H1   = (half_t*)salloc((size_t)BT * FF * sizeof(half_t));
    half_t* W16  = (half_t*)salloc((size_t)E * V   * sizeof(half_t));  // max weight (Wout)
    if (off > ws_size) return;  // insufficient scratch; bail deterministically

    float* logits = (float*)d_out;
    float* loss   = logits + (size_t)BT * V;
    const float scale = 0.125f;  // HS^-0.5 = 1/8
    const long ZB = 0;

    // ---- embedding ----
    embed_kernel<<<BT, 256, 0, stream>>>(idx, tok_emb, pos_emb, X, T, E);

    for (int l = 0; l < L; ++l) {
        // -- attention block --
        layernorm_kernel<<<BT, 256, 0, stream>>>(X, ln1_g + (long)l * E, ln1_b + (long)l * E, H16, E);

        launch_cvt(stream, Wq + (long)l * E * E, W16, (long)E * E);
        launch_gemm(stream, H16, W16, nullptr, nullptr, Q16, BT, E, E, 0,
                    E, E, E, ZB, 0, 0, ZB, 0, 0, ZB, 0, 0, 1, 1, GEMM_OUT_F16);
        launch_cvt(stream, Wk + (long)l * E * E, W16, (long)E * E);
        launch_gemm(stream, H16, W16, nullptr, nullptr, K16, BT, E, E, 0,
                    E, E, E, ZB, 0, 0, ZB, 0, 0, ZB, 0, 0, 1, 1, GEMM_OUT_F16);
        launch_cvt(stream, Wv + (long)l * E * E, W16, (long)E * E);
        launch_gemm(stream, H16, W16, nullptr, nullptr, V16, BT, E, E, 0,
                    E, E, E, ZB, 0, 0, ZB, 0, 0, ZB, 0, 0, 1, 1, GEMM_OUT_F16);

        // scores: per (b,h)  S = Q · K^T   [T x T], K = HS
        launch_gemm(stream, Q16, K16, nullptr, S, nullptr, T, T, HS, /*transB=*/1,
                    E, E, T,
                    ZB, (long)T * E, (long)HS,
                    ZB, (long)T * E, (long)HS,
                    ZB, (long)H * T * T, (long)T * T,
                    Bb, H, GEMM_OUT_F32);

        softmax_kernel<<<Bb * H * T, 256, 0, stream>>>(S, P16, T, scale);

        // context: per (b,h)  O = P · V   [T x HS], K = T
        launch_gemm(stream, P16, V16, nullptr, nullptr, O16, T, HS, T, /*transB=*/0,
                    T, E, E,
                    ZB, (long)H * T * T, (long)T * T,
                    ZB, (long)T * E, (long)HS,
                    ZB, (long)T * E, (long)HS,
                    Bb, H, GEMM_OUT_F16);

        // projection + residual:  X += O · Wo + bo
        launch_cvt(stream, Wo + (long)l * E * E, W16, (long)E * E);
        launch_gemm(stream, O16, W16, bo + (long)l * E, X, nullptr, BT, E, E, 0,
                    E, E, E, ZB, 0, 0, ZB, 0, 0, ZB, 0, 0, 1, 1, GEMM_ACC_F32);

        // -- MLP block --
        layernorm_kernel<<<BT, 256, 0, stream>>>(X, ln2_g + (long)l * E, ln2_b + (long)l * E, H16, E);

        launch_cvt(stream, W1 + (long)l * E * FF, W16, (long)E * FF);
        launch_gemm(stream, H16, W16, b1 + (long)l * FF, nullptr, H1, BT, FF, E, 0,
                    E, FF, FF, ZB, 0, 0, ZB, 0, 0, ZB, 0, 0, 1, 1, GEMM_OUT_F16_RELU);

        launch_cvt(stream, W2 + (long)l * FF * E, W16, (long)FF * E);
        launch_gemm(stream, H1, W16, b2 + (long)l * E, X, nullptr, BT, E, FF, 0,
                    FF, E, E, ZB, 0, 0, ZB, 0, 0, ZB, 0, 0, 1, 1, GEMM_ACC_F32);
    }

    // ---- final LN + logits + loss ----
    layernorm_kernel<<<BT, 256, 0, stream>>>(X, lnf_g, lnf_b, H16, E);

    launch_cvt(stream, Wout, W16, (long)E * V);
    launch_gemm(stream, H16, W16, bout, logits, nullptr, BT, V, E, 0,
                E, V, V, ZB, 0, 0, ZB, 0, 0, ZB, 0, 0, 1, 1, GEMM_OUT_F32);

    zero_kernel<<<1, 1, 0, stream>>>(loss);
    xent_kernel<<<BT, 256, 0, stream>>>(logits, targets, loss, V, 1.0f / BT);
}
